// HGNN_link_prediction_25262997635472
// MI455X (gfx1250) — matrix operand order repaired and checked by reading, not compiled, not run on hardware
//
#include <hip/hip_runtime.h>
#include <hip/hip_bf16.h>
#include <stdint.h>

// ---------------------------------------------------------------------------
// HGNN link prediction, CDNA5 (gfx1250) wave32 + bf16 WMMA implementation.
// N=65536 nodes, D=256, NS=512 slots, HID=256, top-k = 10.
// GEMM tiles staged with GLOBAL_LOAD_ASYNC_TO_LDS_B128 (ASYNCcnt), double-
// buffered LDS (statically unrolled ping/pong), v_wmma_f32_16x16x32_bf16.
// ---------------------------------------------------------------------------

constexpr int kN  = 65536;
constexpr int kD  = 256;
constexpr int kNS = 512;
constexpr float kScale = 0.0625f;   // 256^-0.5
constexpr float kEps   = 1e-8f;

typedef __attribute__((ext_vector_type(16))) __bf16 v16bf;
typedef __attribute__((ext_vector_type(8)))  float  v8f;

union FragB { v16bf v; uint4 q[2]; };

static __device__ __forceinline__ unsigned short f2bf(float x) {
  unsigned int u = __float_as_uint(x);
  unsigned int r = (u + 0x7fffu + ((u >> 16) & 1u)) >> 16;
  return (unsigned short)r;
}

// Low 32 bits of a flat pointer into the LDS aperture are the wave-relative
// LDS byte address (ISA 10.2).
static __device__ __forceinline__ unsigned lds_addr32(const void* p) {
  return (unsigned)(uintptr_t)p;
}

// GLOBAL_LOAD_ASYNC_TO_LDS_B128: per-lane 16B global -> LDS DMA, ASYNCcnt.
static __device__ __forceinline__ void async_ld_b128(unsigned lds, const void* g) {
  unsigned long long ga = (unsigned long long)(uintptr_t)g;
  asm volatile("global_load_async_to_lds_b128 %0, %1, off"
               :: "v"(lds), "v"(ga) : "memory");
}

static __device__ __forceinline__ void wait_async0() {
  asm volatile("s_wait_asynccnt 0x0" ::: "memory");
}

static __device__ __forceinline__ void osm_update(float& m, float& Z, float v) {
  float M = fmaxf(m, v);
  Z = Z * __expf(m - M) + __expf(v - M);
  m = M;
}

static __device__ __forceinline__ void osm_merge(float& m, float& Z, float m2, float Z2) {
  float M = fmaxf(m, m2);
  Z = Z * __expf(m - M) + Z2 * __expf(m2 - M);
  m = M;
}

static __device__ __forceinline__ void topk_insert(float* tv, int* ti, float v, int idx) {
  if (v > tv[9]) {
    tv[9] = v; ti[9] = idx;
    #pragma unroll
    for (int j = 9; j > 0; --j) {
      if (tv[j] > tv[j - 1]) {
        float a = tv[j]; tv[j] = tv[j - 1]; tv[j - 1] = a;
        int   b = ti[j]; ti[j] = ti[j - 1]; ti[j - 1] = b;
      }
    }
  }
}

// ---------------------------------------------------------------------------
// fp32 -> bf16 conversion (weights)
// ---------------------------------------------------------------------------
__global__ void cvt_bf16(const float* __restrict__ in, unsigned short* __restrict__ out, int n) {
  int i = blockIdx.x * 256 + threadIdx.x;
  if (i < n) out[i] = f2bf(in[i]);
}

// ---------------------------------------------------------------------------
// Tiled bf16 WMMA GEMM:  C[M,N] = act( alpha * A[M,K] @ B[K,N] + bias + adds )
// A row stride = lda (elements).  B row stride = N.  Block tile 128x128,
// 256 threads = 8 wave32 waves (4 along M x 2 along N), wave tile 32x64.
// K stepped by 32 (K must be a multiple of 64 -> even tile count); tiles are
// staged via async Global->LDS DMA into statically-unrolled ping/pong buffers.
// ---------------------------------------------------------------------------
__global__ __launch_bounds__(256) void gemm_bf16(
    const unsigned short* __restrict__ A, int lda,
    const unsigned short* __restrict__ B,
    const float* __restrict__ bias, float alpha,
    int M, int N, int K, int relu,
    float* __restrict__ Cf, unsigned short* __restrict__ Cb,
    const float* __restrict__ add0, const float* __restrict__ add1)
{
  __shared__ alignas(16) unsigned short sA[2][128 * 40];   // 128 x 32 halves, pad 40
  __shared__ alignas(16) unsigned short sB[2][32 * 136];   // 32 x 128 halves, pad 136

  const int t    = threadIdx.x;
  const int m0   = blockIdx.y * 128;
  const int n0   = blockIdx.x * 128;
  const int wid  = t >> 5, lane = t & 31;
  const int wm   = wid & 3;        // wave position along M (0..3) -> 32 rows each
  const int wn   = wid >> 2;       // wave position along N (0..1) -> 64 cols each

  // ---- per-thread staging slots: 2 uint4 of A tile + 2 uint4 of B tile ----
  const int f0 = t, f1 = t + 256;
  const int ar0 = f0 >> 2, ac0 = (f0 & 3) << 3;    // A: 4 uint4 per 32-half row
  const int ar1 = f1 >> 2, ac1 = (f1 & 3) << 3;
  const int br0 = f0 >> 4, bc0 = (f0 & 15) << 3;   // B: 16 uint4 per 128-half row
  const int br1 = f1 >> 4, bc1 = (f1 & 15) << 3;

  const unsigned short* aP0 = A + (size_t)(m0 + ar0) * lda + ac0;
  const unsigned short* aP1 = A + (size_t)(m0 + ar1) * lda + ac1;
  const unsigned short* bP0 = B + (size_t)br0 * N + n0 + bc0;
  const unsigned short* bP1 = B + (size_t)br1 * N + n0 + bc1;
  const size_t bStep = (size_t)32 * N;             // halves per K-step for B

  unsigned lA0[2], lA1[2], lB0[2], lB1[2];
  #pragma unroll
  for (int p = 0; p < 2; ++p) {
    lA0[p] = lds_addr32(&sA[p][ar0 * 40 + ac0]);
    lA1[p] = lds_addr32(&sA[p][ar1 * 40 + ac1]);
    lB0[p] = lds_addr32(&sB[p][br0 * 136 + bc0]);
    lB1[p] = lds_addr32(&sB[p][br1 * 136 + bc1]);
  }

  v8f acc[2][4];
  #pragma unroll
  for (int mi = 0; mi < 2; ++mi)
    #pragma unroll
    for (int ni = 0; ni < 4; ++ni)
      #pragma unroll
      for (int i = 0; i < 8; ++i) acc[mi][ni][i] = 0.f;

  const int arow = wm * 32 + (lane & 15);
  const int koff = (lane >> 4) << 3;   // lanes 0-15: K 0-7/16-23; 16-31: 8-15/24-31

  // stage one 32-deep K tile into ping/pong buffer p (async DMA, ASYNCcnt)
  auto stage = [&](int p) {
    async_ld_b128(lA0[p], aP0);
    async_ld_b128(lA1[p], aP1);
    async_ld_b128(lB0[p], bP0);
    async_ld_b128(lB1[p], bP1);
    aP0 += 32; aP1 += 32; bP0 += bStep; bP1 += bStep;
  };

  // consume buffer p: load fragments (wave32 ISA layouts, §7.12.2) + 8 WMMAs
  auto compute = [&](int p) {
    const unsigned short* cA = sA[p];
    const unsigned short* cB = sB[p];
    FragB a[2], b[4];
    #pragma unroll
    for (int mi = 0; mi < 2; ++mi) {
      int rr = arow + mi * 16;
      a[mi].q[0] = *(const uint4*)&cA[rr * 40 + koff];
      a[mi].q[1] = *(const uint4*)&cA[rr * 40 + 16 + koff];
    }
    #pragma unroll
    for (int ni = 0; ni < 4; ++ni) {
      int cc = wn * 64 + ni * 16 + (lane & 15);      // B: column = lane%16, K as A
      #pragma unroll
      for (int s = 0; s < 8; ++s) {
        b[ni].v[s]     = __builtin_bit_cast(__bf16, cB[(koff + s) * 136 + cc]);
        b[ni].v[8 + s] = __builtin_bit_cast(__bf16, cB[(16 + koff + s) * 136 + cc]);
      }
    }
    #pragma unroll
    for (int mi = 0; mi < 2; ++mi)
      #pragma unroll
      for (int ni = 0; ni < 4; ++ni)
        acc[mi][ni] = __builtin_amdgcn_wmma_f32_16x16x32_bf16(
            false, a[mi].v, false, b[ni].v, (short)0, acc[mi][ni], false, false);
  };

  const int nk = K >> 5;               // even for all GEMMs in this pipeline
  stage(0);                            // prologue: tile 0 -> buffer 0
  for (int kt = 0; kt < nk; kt += 2) {
    wait_async0();                     // tile kt resident in buffer 0
    __syncthreads();                   // everyone done reading buffer 1
    stage(1);                          // tile kt+1 -> buffer 1 (nk even)
    compute(0);
    wait_async0();                     // tile kt+1 resident in buffer 1
    __syncthreads();                   // everyone done reading buffer 0
    if (kt + 2 < nk) stage(0);         // tile kt+2 -> buffer 0
    compute(1);
  }

  // ---- epilogue: C/D layout — VGPR i: lanes 0-15 row i, lanes 16-31 row i+8
  const int crBase = m0 + wm * 32;
  const int ccBase = n0 + wn * 64;
  const int colw   = lane & 15;
  const int hi     = (lane >> 4) << 3;
  #pragma unroll
  for (int mi = 0; mi < 2; ++mi) {
    #pragma unroll
    for (int ni = 0; ni < 4; ++ni) {
      const int cc = ccBase + ni * 16 + colw;
      const float bv = bias ? bias[cc] : 0.f;
      #pragma unroll
      for (int i = 0; i < 8; ++i) {
        const int cr = crBase + mi * 16 + i + hi;
        const size_t off = (size_t)cr * N + cc;
        float v = acc[mi][ni][i] * alpha + bv;
        if (add0) v += add0[off];
        if (add1) v += add1[off];
        if (relu) v = fmaxf(v, 0.f);
        if (Cf) Cf[off] = v;
        if (Cb) Cb[off] = f2bf(v);
      }
    }
  }
}

// ---------------------------------------------------------------------------
// edges = mu + exp(logsigma)*noise ; e = LN(edges) -> cat_bf[:, 0:256] (bf16)
// one block per slot row
// ---------------------------------------------------------------------------
__global__ __launch_bounds__(256) void edges_ln(
    const float* __restrict__ noise, const float* __restrict__ mu,
    const float* __restrict__ logsig, const float* __restrict__ w,
    const float* __restrict__ b, unsigned short* __restrict__ cat_bf)
{
  const int s = blockIdx.x, d = threadIdx.x;
  float v = mu[d] + __expf(logsig[d]) * noise[s * kD + d];
  __shared__ float red[256];
  red[d] = v; __syncthreads();
  for (int o = 128; o > 0; o >>= 1) { if (d < o) red[d] += red[d + o]; __syncthreads(); }
  float mean = red[0] * (1.f / kD);
  __syncthreads();
  float c = v - mean;
  red[d] = c * c; __syncthreads();
  for (int o = 128; o > 0; o >>= 1) { if (d < o) red[d] += red[d + o]; __syncthreads(); }
  float var = red[0] * (1.f / kD);
  float o1 = c * rsqrtf(var + 1e-5f) * w[d] + b[d];
  cat_bf[s * (2 * kD) + d] = f2bf(o1);
}

// ---------------------------------------------------------------------------
// xin = LN(x) -> bf16 ; x -> bf16.  One wave per row, 8 rows per block.
// ---------------------------------------------------------------------------
__global__ __launch_bounds__(256) void ln_x(
    const float* __restrict__ x, const float* __restrict__ w, const float* __restrict__ b,
    unsigned short* __restrict__ xin_bf, unsigned short* __restrict__ x_bf)
{
  const int wid = threadIdx.x >> 5, lane = threadIdx.x & 31;
  const int r = blockIdx.x * 8 + wid;
  const float* xr = x + (size_t)r * kD;
  float4 a = *(const float4*)(xr + lane * 4);
  float4 c = *(const float4*)(xr + 128 + lane * 4);
  float s1 = a.x + a.y + a.z + a.w + c.x + c.y + c.z + c.w;
  float s2 = a.x * a.x + a.y * a.y + a.z * a.z + a.w * a.w +
             c.x * c.x + c.y * c.y + c.z * c.z + c.w * c.w;
  #pragma unroll
  for (int o = 16; o > 0; o >>= 1) {
    s1 += __shfl_xor(s1, o, 32);
    s2 += __shfl_xor(s2, o, 32);
  }
  float mean = s1 * (1.f / kD);
  float var  = s2 * (1.f / kD) - mean * mean;
  float rstd = rsqrtf(var + 1e-5f);
  float va[8] = {a.x, a.y, a.z, a.w, c.x, c.y, c.z, c.w};
  size_t rb = (size_t)r * kD;
  #pragma unroll
  for (int j = 0; j < 8; ++j) {
    int col = (j < 4) ? (lane * 4 + j) : (128 + lane * 4 + (j - 4));
    float o1 = (va[j] - mean) * rstd * w[col] + b[col];
    xin_bf[rb + col] = f2bf(o1);
    x_bf[rb + col]   = f2bf(va[j]);
  }
}

// ---------------------------------------------------------------------------
// bf16 transpose [R,C] -> [C,R] (small matrices only)
// ---------------------------------------------------------------------------
__global__ void transpose_bf(const unsigned short* __restrict__ in,
                             unsigned short* __restrict__ out, int R, int C)
{
  int r = blockIdx.x;
  for (int c = threadIdx.x; c < C; c += blockDim.x)
    out[(size_t)c * R + r] = in[(size_t)r * C + c];
}

// ---------------------------------------------------------------------------
// Slot attention reduce: dots[r,s] over r (65536), per-slot online softmax +
// top-10; updates[s] = sum_top10 w_i * vv[idx_i,:] -> cat_bf[:, 256:512].
// One block per slot; dots is L2-resident (128 MB < 192 MB L2).
// ---------------------------------------------------------------------------
__global__ __launch_bounds__(256) void slot_attn(
    const float* __restrict__ dots, const float* __restrict__ vv,
    unsigned short* __restrict__ cat_bf)
{
  const int s = blockIdx.x, t = threadIdx.x;
  float m = -3.0e38f, Z = 0.f;
  float tv[10]; int ti[10];
  #pragma unroll
  for (int i = 0; i < 10; ++i) { tv[i] = -3.0e38f; ti[i] = 0; }
  for (int r = t; r < kN; r += 256) {
    float v = dots[(size_t)r * kNS + s];
    osm_update(m, Z, v);
    topk_insert(tv, ti, v, r);
  }
  __shared__ float sm[256], sz[256];
  sm[t] = m; sz[t] = Z; __syncthreads();
  for (int o = 128; o > 0; o >>= 1) {
    if (t < o) {
      float mm = sm[t], zz = sz[t];
      osm_merge(mm, zz, sm[t + o], sz[t + o]);
      sm[t] = mm; sz[t] = zz;
    }
    __syncthreads();
  }
  __shared__ float svv[2560];
  __shared__ int   sii[2560];
  #pragma unroll
  for (int i = 0; i < 10; ++i) { svv[t * 10 + i] = tv[i]; sii[t * 10 + i] = ti[i]; }
  __shared__ float wsel[10];
  __shared__ int   isel[10];
  __syncthreads();
  if (t == 0) {
    const float M = sm[0], Zf = sz[0];
    const float inv = 1.0f / (1.0f + (float)kN * kEps);
    for (int k = 0; k < 10; ++k) {
      float bv = -3.0e38f; int bj = 0;
      for (int j = 0; j < 2560; ++j) if (svv[j] > bv) { bv = svv[j]; bj = j; }
      wsel[k] = (__expf(bv - M) / Zf + kEps) * inv;
      isel[k] = sii[bj];
      svv[bj] = -3.0e38f;
    }
  }
  __syncthreads();
  float acc = 0.f;
  #pragma unroll
  for (int k = 0; k < 10; ++k) acc += wsel[k] * vv[(size_t)isel[k] * kD + t];
  cat_bf[s * (2 * kD) + kD + t] = f2bf(acc);
}

// ---------------------------------------------------------------------------
// Node attention: per node row softmax over 512 + top-10;
// nodes[r] = sum_top10 p_i * edg[idx_i,:].  One wave per row.
// ---------------------------------------------------------------------------
__global__ __launch_bounds__(256) void node_attn(
    const float* __restrict__ dots2, const float* __restrict__ edg,
    float* __restrict__ nodes_f, unsigned short* __restrict__ nodes_bf)
{
  const int wid = threadIdx.x >> 5, lane = threadIdx.x & 31;
  const int r = blockIdx.x * 8 + wid;
  const float* dr = dots2 + (size_t)r * kNS;
  float m = -3.0e38f, Z = 0.f;
  float tv[10]; int ti[10];
  #pragma unroll
  for (int i = 0; i < 10; ++i) { tv[i] = -3.0e38f; ti[i] = 0; }
  #pragma unroll
  for (int j = 0; j < 4; ++j) {
    float4 q = *(const float4*)(dr + j * 128 + lane * 4);
    int cbase = j * 128 + lane * 4;
    float vv4[4] = {q.x, q.y, q.z, q.w};
    #pragma unroll
    for (int c = 0; c < 4; ++c) { osm_update(m, Z, vv4[c]); topk_insert(tv, ti, vv4[c], cbase + c); }
  }
  #pragma unroll
  for (int o = 16; o > 0; o >>= 1) {
    float m2 = __shfl_xor(m, o, 32);
    float z2 = __shfl_xor(Z, o, 32);
    osm_merge(m, Z, m2, z2);
  }
  __shared__ float sv[8][320];
  __shared__ int   si[8][320];
  __shared__ float wsel[8][10];
  __shared__ int   isel[8][10];
  #pragma unroll
  for (int i = 0; i < 10; ++i) { sv[wid][lane * 10 + i] = tv[i]; si[wid][lane * 10 + i] = ti[i]; }
  __syncthreads();
  if (lane == 0) {
    for (int k = 0; k < 10; ++k) {
      float bv = -3.0e38f; int bj = 0;
      for (int j = 0; j < 320; ++j) if (sv[wid][j] > bv) { bv = sv[wid][j]; bj = j; }
      wsel[wid][k] = __expf(bv - m) / Z;
      isel[wid][k] = si[wid][bj];
      sv[wid][bj] = -3.0e38f;
    }
  }
  __syncthreads();
  float acc[8];
  #pragma unroll
  for (int j = 0; j < 8; ++j) acc[j] = 0.f;
  #pragma unroll
  for (int k = 0; k < 10; ++k) {
    float p = wsel[wid][k];
    const float* er = edg + (size_t)isel[wid][k] * kD;
    #pragma unroll
    for (int j = 0; j < 8; ++j) acc[j] += p * er[lane + 32 * j];
  }
  size_t rb = (size_t)r * kD;
  #pragma unroll
  for (int j = 0; j < 8; ++j) {
    nodes_f[rb + lane + 32 * j]  = acc[j];
    nodes_bf[rb + lane + 32 * j] = f2bf(acc[j]);
  }
}

// ---------------------------------------------------------------------------
// Host launcher
// ---------------------------------------------------------------------------
extern "C" void kernel_launch(void* const* d_in, const int* in_sizes, int n_in,
                              void* d_out, int out_size, void* d_ws, size_t ws_size,
                              hipStream_t stream) {
  (void)in_sizes; (void)n_in; (void)out_size; (void)ws_size;

  const float* x        = (const float*)d_in[0];
  const float* e_noise  = (const float*)d_in[1];
  const float* e_mu     = (const float*)d_in[2];
  const float* e_logsig = (const float*)d_in[3];
  const float* ln_in_w  = (const float*)d_in[4];
  const float* ln_in_b  = (const float*)d_in[5];
  const float* ln_e_w   = (const float*)d_in[6];
  const float* ln_e_b   = (const float*)d_in[7];
  const float* Wq  = (const float*)d_in[8];   const float* bq   = (const float*)d_in[9];
  const float* Wk  = (const float*)d_in[10];  const float* bk   = (const float*)d_in[11];
  const float* Wv  = (const float*)d_in[12];  const float* bv   = (const float*)d_in[13];
  const float* Wm1 = (const float*)d_in[14];  const float* bm1  = (const float*)d_in[15];
  const float* Wm2 = (const float*)d_in[16];  const float* bm2  = (const float*)d_in[17];
  const float* Wcv = (const float*)d_in[18];  const float* bcv  = (const float*)d_in[19];
  const float* Wc0 = (const float*)d_in[20];  const float* bc0  = (const float*)d_in[21];
  const float* Wc1 = (const float*)d_in[22];  const float* bc1  = (const float*)d_in[23];
  const float* Wt  = (const float*)d_in[24];  const float* bt   = (const float*)d_in[25];
  float* out = (float*)d_out;

  // ---- workspace carve ----
  char* base = (char*)d_ws;
  size_t off = 0;
  auto alloc = [&](size_t bytes) -> void* {
    off = (off + 255) & ~(size_t)255;
    void* p = base + off;
    off += bytes;
    return p;
  };
  typedef unsigned short u16;
  u16*   cat_bf    = (u16*)alloc((size_t)kNS * 2 * kD * 2);      // [512, 512] bf16 (e | updates)
  u16*   xin_bf    = (u16*)alloc((size_t)kN * kD * 2);
  u16*   x_bf      = (u16*)alloc((size_t)kN * kD * 2);
  u16*   kk_bf     = (u16*)alloc((size_t)kN * kD * 2);
  u16*   q2_bf     = (u16*)alloc((size_t)kN * kD * 2);
  float* vv_f      = (float*)alloc((size_t)kN * kD * 4);
  float* dots_f    = (float*)alloc((size_t)kN * kNS * 4);        // reused for dots1^T and dots2
  float* nodes_f   = (float*)alloc((size_t)kN * kD * 4);
  u16*   nodes_bf  = (u16*)alloc((size_t)kN * kD * 2);
  float* t1_f      = (float*)alloc((size_t)kN * kD * 4);
  u16*   outpre_bf = (u16*)alloc((size_t)kN * kD * 2);
  u16*   q_bf      = (u16*)alloc((size_t)kNS * kD * 2);
  u16*   qT_bf     = (u16*)alloc((size_t)kNS * kD * 2);
  u16*   h_bf      = (u16*)alloc((size_t)kNS * kD * 2);
  u16*   e2_bf     = (u16*)alloc((size_t)kNS * kD * 2);
  u16*   k2_bf     = (u16*)alloc((size_t)kNS * kD * 2);
  u16*   k2T_bf    = (u16*)alloc((size_t)kNS * kD * 2);
  float* edg_f     = (float*)alloc((size_t)kNS * kD * 4);
  u16* WqB  = (u16*)alloc((size_t)kD * kD * 2);
  u16* WkB  = (u16*)alloc((size_t)kD * kD * 2);
  u16* WvB  = (u16*)alloc((size_t)kD * kD * 2);
  u16* Wm1B = (u16*)alloc((size_t)2 * kD * kD * 2);
  u16* Wm2B = (u16*)alloc((size_t)kD * kD * 2);
  u16* WcvB = (u16*)alloc((size_t)kD * kD * 2);
  u16* Wc0B = (u16*)alloc((size_t)kD * kD * 2);
  u16* Wc1B = (u16*)alloc((size_t)kD * kD * 2);
  u16* WtB  = (u16*)alloc((size_t)kD * kD * 2);

  const dim3 blk(256);
  auto cvt = [&](const float* in, u16* o, int n) {
    cvt_bf16<<<dim3((n + 255) / 256), blk, 0, stream>>>(in, o, n);
  };
  auto gemm = [&](const u16* A, int lda, const u16* B, const float* bias, float alpha,
                  int M, int N, int K, int relu, float* Cf, u16* Cb,
                  const float* a0, const float* a1) {
    gemm_bf16<<<dim3(N / 128, M / 128), blk, 0, stream>>>(
        A, lda, B, bias, alpha, M, N, K, relu, Cf, Cb, a0, a1);
  };

  // ---- weight conversions ----
  cvt(Wq, WqB, kD * kD);   cvt(Wk, WkB, kD * kD);   cvt(Wv, WvB, kD * kD);
  cvt(Wm1, Wm1B, 2 * kD * kD); cvt(Wm2, Wm2B, kD * kD); cvt(Wcv, WcvB, kD * kD);
  cvt(Wc0, Wc0B, kD * kD); cvt(Wc1, Wc1B, kD * kD); cvt(Wt, WtB, kD * kD);

  // ---- small setup ----
  edges_ln<<<dim3(kNS), blk, 0, stream>>>(e_noise, e_mu, e_logsig, ln_e_w, ln_e_b, cat_bf);
  ln_x<<<dim3(kN / 8), blk, 0, stream>>>(x, ln_in_w, ln_in_b, xin_bf, x_bf);

  // ---- q = relu(e @ Wq + bq) ; q^T for the score GEMM ----
  gemm(cat_bf, 2 * kD, WqB, bq, 1.f, kNS, kD, kD, 1, nullptr, q_bf, nullptr, nullptr);
  transpose_bf<<<dim3(kNS), blk, 0, stream>>>(q_bf, qT_bf, kNS, kD);

  // ---- node-side projections ----
  gemm(xin_bf, kD, WkB, bk, 1.f, kN, kD, kD, 1, nullptr, kk_bf, nullptr, nullptr); // kk
  gemm(xin_bf, kD, WvB, bv, 1.f, kN, kD, kD, 1, vv_f, nullptr, nullptr, nullptr);  // vv
  gemm(xin_bf, kD, WqB, bq, 1.f, kN, kD, kD, 0, nullptr, q2_bf, nullptr, nullptr); // q2 (no relu)

  // ---- dots1^T = (kk @ q^T) * scale   [65536, 512] ----
  gemm(kk_bf, kD, qT_bf, nullptr, kScale, kN, kNS, kD, 0, dots_f, nullptr, nullptr, nullptr);

  // ---- per-slot softmax + top-10 + sparse updates -> cat_bf[:,256:512] ----
  slot_attn<<<dim3(kNS), blk, 0, stream>>>(dots_f, vv_f, cat_bf);

  // ---- slot MLP: e2 = relu(cat @ Wm1 + bm1) @ Wm2 + bm2 ----
  gemm(cat_bf, 2 * kD, Wm1B, bm1, 1.f, kNS, kD, 2 * kD, 1, nullptr, h_bf, nullptr, nullptr);
  gemm(h_bf, kD, Wm2B, bm2, 1.f, kNS, kD, kD, 0, nullptr, e2_bf, nullptr, nullptr);

  // ---- k2 = relu(e2 @ Wk + bk) ; k2^T ; edg = e2 @ conv_w + conv_b ----
  gemm(e2_bf, kD, WkB, bk, 1.f, kNS, kD, kD, 1, nullptr, k2_bf, nullptr, nullptr);
  transpose_bf<<<dim3(kNS), blk, 0, stream>>>(k2_bf, k2T_bf, kNS, kD);
  gemm(e2_bf, kD, WcvB, bcv, 1.f, kNS, kD, kD, 0, edg_f, nullptr, nullptr, nullptr);

  // ---- dots2 = (q2 @ k2^T) * scale   [65536, 512] (reuse dots buffer) ----
  gemm(q2_bf, kD, k2T_bf, nullptr, kScale, kN, kNS, kD, 0, dots_f, nullptr, nullptr, nullptr);

  // ---- per-node softmax + top-10 + sparse gather -> nodes ----
  node_attn<<<dim3(kN / 8), blk, 0, stream>>>(dots_f, edg_f, nodes_f, nodes_bf);

  // ---- out = relu(x@Wc0+bc0 + nodes@Wc1+bc1 + nodes) ; out = relu(out@Wt+bt) ----
  gemm(x_bf, kD, Wc0B, bc0, 1.f, kN, kD, kD, 0, t1_f, nullptr, nullptr, nullptr);
  gemm(nodes_bf, kD, Wc1B, bc1, 1.f, kN, kD, kD, 1, nullptr, outpre_bf, t1_f, nodes_f);
  gemm(outpre_bf, kD, WtB, bt, 1.f, kN, kD, kD, 1, out, nullptr, nullptr, nullptr);
}
